// ResNet_smpl_87617332838551
// MI455X (gfx1250) — matrix operand order repaired and checked by reading, not compiled, not run on hardware
//
#include <hip/hip_runtime.h>

// ---------------------------------------------------------------------------
// CDNA5 / gfx1250 ResNet18-backbone + dual SMPL.
// Activations NHWC f16: implicit-GEMM K-chunks (32 input channels at a fixed
// filter tap) are contiguous -> b128 loads feeding v_wmma_f32_16x16x32_f16.
// Each wave owns a 32x64 output macro-tile (8 accumulators, 4x A-frag reuse,
// 2x B-frag reuse). BN/residual/ReLU fused in epilogue.
// ---------------------------------------------------------------------------

typedef __attribute__((ext_vector_type(16))) _Float16 v16h;
typedef __attribute__((ext_vector_type(8)))  _Float16 v8h;
typedef __attribute__((ext_vector_type(8)))  float    v8f;

#define SMPL_V  6890
#define SMPL_NJ 24
#define BATCH   64

// ---------------------------------------------------------------------------
// prep kernels
// ---------------------------------------------------------------------------
__global__ void k_f32_to_f16(const float* __restrict__ s, _Float16* __restrict__ d, int n) {
  int i = blockIdx.x * blockDim.x + threadIdx.x;
  if (i < n) d[i] = (_Float16)s[i];
}

// OIHW f32 -> [tap][co][ci] f16  (tap = ky*KW+kx); makes B-fragment K contiguous
__global__ void k_pack_wgt(const float* __restrict__ src, _Float16* __restrict__ dst,
                           int Co, int Ci, int KH, int KW) {
  int i = blockIdx.x * blockDim.x + threadIdx.x;
  int total = Co * Ci * KH * KW;
  if (i >= total) return;
  int kx = i % KW; int t = i / KW; int ky = t % KH; t /= KH; int ci = t % Ci; int co = t / Ci;
  dst[((long)(ky * KW + kx) * Co + co) * Ci + ci] = (_Float16)src[i];
}

__global__ void k_bn_prep(const float* __restrict__ g, const float* __restrict__ b,
                          const float* __restrict__ m, const float* __restrict__ v,
                          float* __restrict__ scale, float* __restrict__ shift, int C) {
  int c = blockIdx.x * blockDim.x + threadIdx.x;
  if (c < C) {
    float sc = g[c] * rsqrtf(v[c] + 1e-5f);
    scale[c] = sc;
    shift[c] = b[c] - m[c] * sc;
  }
}

// ---------------------------------------------------------------------------
// Fast NHWC implicit-GEMM conv (Ci mult of 32; Co mult of 64; M mult of 32).
// Wave32 per block; wave owns 32(M) x 64(N) outputs = 8 WMMA accumulators.
// A frag (16x32): lane l (half=l>>4, m=l&15) holds K = half*16+0..15
//   -> two contiguous v8h (b128) loads from NHWC at fixed tap.
// B frag (32x16): lane holds K = half*16+0..15 -> one contiguous v16h load from
//   packed [tap][co][ci] weights.
// C/D: VGPR r, lane l -> (m = r + 8*half, n = l&15).
// ---------------------------------------------------------------------------
__global__ __launch_bounds__(32)
void k_conv_wmma_nhwc(const _Float16* __restrict__ in, const _Float16* __restrict__ wgt,
                      const float* __restrict__ scale, const float* __restrict__ shift,
                      const _Float16* __restrict__ res, _Float16* __restrict__ out,
                      int B, int Ci, int Hin, int Win, int Co, int H, int W,
                      int KH, int KW, int stride, int pad, int relu) {
  const int lane = threadIdx.x;
  const int half = lane >> 4, l15 = lane & 15;
  const int HW = H * W;
  const int taps = KH * KW;

  // decode the two 16-pixel M-subtiles this wave covers (A rows)
  int oy[2], ox[2], bb[2];
#pragma unroll
  for (int s = 0; s < 2; ++s) {
    int m = blockIdx.x * 32 + s * 16 + l15;
    bb[s] = m / HW; int r = m % HW; oy[s] = r / W; ox[s] = r % W;
  }
  const int co0 = blockIdx.y * 64 + l15;   // N-subtiles at co0 + {0,16,32,48}

  v8f acc[2][4] = {{{}, {}, {}, {}}, {{}, {}, {}, {}}};

  for (int tap = 0; tap < taps; ++tap) {
    const int ky = tap / KW, kx = tap % KW;
    long abase[2]; bool ok[2];
#pragma unroll
    for (int s = 0; s < 2; ++s) {
      int iy = oy[s] * stride - pad + ky;
      int ix = ox[s] * stride - pad + kx;
      ok[s] = (iy >= 0 && iy < Hin && ix >= 0 && ix < Win);
      abase[s] = (((long)bb[s] * Hin + iy) * Win + ix) * Ci;
    }
    const _Float16* wb = wgt + ((long)tap * Co + co0) * Ci;
    if (tap + 1 < taps)                      // pull next tap's weights toward L2/L0
      __builtin_prefetch(wb + (long)Co * Ci, 0, 1);

#pragma unroll 2
    for (int kb = 0; kb < Ci; kb += 32) {
      v16h af[2], bf[4];
#pragma unroll
      for (int s = 0; s < 2; ++s) {
        v8h z = {};
        v8h lo = ok[s] ? *(const v8h*)(in + abase[s] + kb + half * 16)      : z;
        v8h hi = ok[s] ? *(const v8h*)(in + abase[s] + kb + half * 16 + 8) : z;
        af[s] = __builtin_shufflevector(lo, hi, 0, 1, 2, 3, 4, 5, 6, 7,
                                                8, 9, 10, 11, 12, 13, 14, 15);
      }
#pragma unroll
      for (int n = 0; n < 4; ++n)
        bf[n] = *(const v16h*)(wb + (long)n * 16 * Ci + kb + half * 16);
#pragma unroll
      for (int s = 0; s < 2; ++s)
#pragma unroll
        for (int n = 0; n < 4; ++n)
          acc[s][n] = __builtin_amdgcn_wmma_f32_16x16x32_f16(
              false, af[s], false, bf[n], (short)0, acc[s][n], false, false);
    }
  }

  // fused BN (+residual) (+ReLU) epilogue, NHWC store
  float sc[4], sh[4];
#pragma unroll
  for (int n = 0; n < 4; ++n) { sc[n] = scale[co0 + n * 16]; sh[n] = shift[co0 + n * 16]; }
#pragma unroll
  for (int s = 0; s < 2; ++s) {
#pragma unroll
    for (int r = 0; r < 8; ++r) {
      int mm = blockIdx.x * 32 + s * 16 + r + 8 * half;
      int b2 = mm / HW; int rr = mm % HW; int yy = rr / W; int xx = rr % W;
      long obase = (((long)b2 * H + yy) * W + xx) * Co + co0;
#pragma unroll
      for (int n = 0; n < 4; ++n) {
        long o = obase + n * 16;
        float val = acc[s][n][r] * sc[n] + sh[n];
        if (res)  val += (float)res[o];
        if (relu) val = val > 0.0f ? val : 0.0f;
        out[o] = (_Float16)val;
      }
    }
  }
}

// ---------------------------------------------------------------------------
// Generic gather-based WMMA conv (used only for conv1: 7x7, Ci=1, K=49).
// NHWC in/out; weights flat [co][K], K = (ci*KH+ky)*KW+kx (OIHW flatten).
// ---------------------------------------------------------------------------
__global__ __launch_bounds__(32)
void k_conv_wmma_gen(const _Float16* __restrict__ in, const _Float16* __restrict__ wgt,
                     const float* __restrict__ scale, const float* __restrict__ shift,
                     _Float16* __restrict__ out,
                     int B, int Ci, int Hin, int Win, int Co, int H, int W,
                     int KH, int KW, int stride, int pad, int relu) {
  const int lane = threadIdx.x;
  const int half = lane >> 4, l15 = lane & 15;
  const int HW = H * W, M = B * HW;
  const int KHW = KH * KW, K = Ci * KHW;
  const int Kpad = (K + 31) & ~31;

  const int m = blockIdx.x * 16 + l15;
  int b = 0, oy = 0, ox = 0;
  const bool mok = (m < M);
  if (mok) { b = m / HW; int r = m % HW; oy = r / W; ox = r % W; }
  const int co = blockIdx.y * 16 + l15;
  const bool nok = (co < Co);

  v8f acc = {};
  for (int kb = 0; kb < Kpad; kb += 32) {
    v16h af, bf;
#pragma unroll
    for (int v = 0; v < 8; ++v) {
#pragma unroll
      for (int e = 0; e < 2; ++e) {
        int ka = kb + ((v & 4) ? 16 : 0) + half * 8 + (v & 3) * 2 + e;
        _Float16 av = (_Float16)0.0f;
        if (mok && ka < K) {
          int ci = ka / KHW; int r = ka % KHW; int ky = r / KW; int kx = r % KW;
          int iy = oy * stride - pad + ky;
          int ix = ox * stride - pad + kx;
          if (iy >= 0 && iy < Hin && ix >= 0 && ix < Win)
            av = in[(((long)b * Hin + iy) * Win + ix) * Ci + ci];
        }
        af[v * 2 + e] = av;
        int kw_ = kb + ((v & 4) ? 16 : 0) + half * 8 + (v & 3) * 2 + e;
        _Float16 bv = (_Float16)0.0f;
        if (nok && kw_ < K) bv = wgt[(long)co * K + kw_];
        bf[v * 2 + e] = bv;
      }
    }
    acc = __builtin_amdgcn_wmma_f32_16x16x32_f16(false, af, false, bf,
                                                 (short)0, acc, false, false);
  }

  if (!nok) return;
  const float sc = scale[co], sh = shift[co];
#pragma unroll
  for (int r = 0; r < 8; ++r) {
    int mm = blockIdx.x * 16 + r + 8 * half;
    if (mm >= M) continue;
    int b2 = mm / HW; int rr = mm % HW; int yy = rr / W; int xx = rr % W;
    long o = (((long)b2 * H + yy) * W + xx) * Co + co;
    float val = acc[r] * sc + sh;
    if (relu) val = val > 0.0f ? val : 0.0f;
    out[o] = (_Float16)val;
  }
}

// ---------------------------------------------------------------------------
// 3x3 stride-2 pad-1 maxpool, NHWC f16
// ---------------------------------------------------------------------------
__global__ void k_maxpool(const _Float16* __restrict__ in, _Float16* __restrict__ out,
                          int B, int C, int Hin, int Win, int H, int W) {
  int i = blockIdx.x * blockDim.x + threadIdx.x;
  int total = B * C * H * W;
  if (i >= total) return;
  int c = i % C; int t = i / C; int xx = t % W; t /= W; int yy = t % H; int bb = t / H;
  float mx = -1e30f;
  for (int ky = 0; ky < 3; ++ky)
    for (int kx = 0; kx < 3; ++kx) {
      int iy = yy * 2 - 1 + ky, ix = xx * 2 - 1 + kx;
      if (iy >= 0 && iy < Hin && ix >= 0 && ix < Win) {
        float v = (float)in[(((long)bb * Hin + iy) * Win + ix) * C + c];
        mx = v > mx ? v : mx;
      }
    }
  out[i] = (_Float16)mx;
}

__global__ void k_avgpool(const _Float16* __restrict__ in, float* __restrict__ feat,
                          int B, int C, int HW) {
  int i = blockIdx.x * blockDim.x + threadIdx.x;
  if (i >= B * C) return;
  int c = i % C, bb = i / C;
  float s = 0.0f;
  for (int p = 0; p < HW; ++p) s += (float)in[((long)bb * HW + p) * C + c];
  feat[i] = s / (float)HW;
}

__global__ void k_fc(const float* __restrict__ feat, const float* __restrict__ wf,
                     const float* __restrict__ bf, float* __restrict__ out,
                     int B, int F, int O) {
  int i = blockIdx.x * blockDim.x + threadIdx.x;
  if (i >= B * O) return;
  int o = i % O, bb = i / O;
  float s = bf[o];
  const float* x = feat + (long)bb * F;
  const float* w = wf + (long)o * F;
  for (int f = 0; f < F; ++f) s += x[f] * w[f];
  out[i] = s;
}

// ---------------------------------------------------------------------------
// SMPL kernels (f32; ~1% of total FLOPs, bandwidth-trivial on 23.3 TB/s HBM)
// ---------------------------------------------------------------------------
__global__ void k_smpl_shape(const float* __restrict__ shapedirs,
                             const float* __restrict__ v_template,
                             const float* __restrict__ feat82,
                             float* __restrict__ vsh) {
  int i = blockIdx.x * blockDim.x + threadIdx.x;
  if (i >= BATCH * SMPL_V) return;
  int v = i % SMPL_V, bb = i / SMPL_V;
  const float* betas = feat82 + bb * 82;
  for (int c = 0; c < 3; ++c) {
    float s = v_template[v * 3 + c];
    const float* sd = shapedirs + ((long)v * 3 + c) * 10;
    for (int k = 0; k < 10; ++k) s += sd[k] * betas[k];
    vsh[(long)i * 3 + c] = s;
  }
}

__global__ __launch_bounds__(256)
void k_jreg(const float* __restrict__ Jr, const float* __restrict__ verts,
            float* __restrict__ J) {
  __shared__ float red[768];
  int bj = blockIdx.x;
  int j = bj % SMPL_NJ, bb = bj / SMPL_NJ;
  int tid = threadIdx.x;
  float s0 = 0, s1 = 0, s2 = 0;
  for (int v = tid; v < SMPL_V; v += 256) {
    float w = Jr[(long)j * SMPL_V + v];
    const float* p = verts + ((long)bb * SMPL_V + v) * 3;
    s0 += w * p[0]; s1 += w * p[1]; s2 += w * p[2];
  }
  red[tid] = s0; red[256 + tid] = s1; red[512 + tid] = s2;
  __syncthreads();
  for (int off = 128; off; off >>= 1) {
    if (tid < off) {
      red[tid] += red[tid + off];
      red[256 + tid] += red[256 + tid + off];
      red[512 + tid] += red[512 + tid + off];
    }
    __syncthreads();
  }
  if (tid == 0) {
    float* o = J + (long)bj * 3;
    o[0] = red[0]; o[1] = red[256]; o[2] = red[512];
  }
}

__global__ void k_rodrigues(const float* __restrict__ feat82,
                            float* __restrict__ Rbuf, float* __restrict__ lrot) {
  int i = blockIdx.x * blockDim.x + threadIdx.x;
  if (i >= BATCH * SMPL_NJ) return;
  int j = i % SMPL_NJ, bb = i / SMPL_NJ;
  const float* r = feat82 + bb * 82 + 10 + j * 3;
  float th = sqrtf(r[0]*r[0] + r[1]*r[1] + r[2]*r[2] + 1e-8f);
  float ax = r[0]/th, ay = r[1]/th, az = r[2]/th;
  float ct = cosf(th), st = sinf(th), o = 1.0f - ct;
  float R[9];
  R[0] = ct + o*ax*ax;    R[1] = o*ax*ay - st*az; R[2] = o*ax*az + st*ay;
  R[3] = o*ay*ax + st*az; R[4] = ct + o*ay*ay;    R[5] = o*ay*az - st*ax;
  R[6] = o*az*ax - st*ay; R[7] = o*az*ay + st*ax; R[8] = ct + o*az*az;
  float* d = Rbuf + (long)i * 9;
  for (int t = 0; t < 9; ++t) d[t] = R[t];
  if (j >= 1) {
    float* lr = lrot + (long)bb * 207 + (j - 1) * 9;
    lr[0] = R[0]-1.f; lr[1] = R[1];     lr[2] = R[2];
    lr[3] = R[3];     lr[4] = R[4]-1.f; lr[5] = R[5];
    lr[6] = R[6];     lr[7] = R[7];     lr[8] = R[8]-1.f;
  }
}

__global__ void k_chain(const float* __restrict__ Rb, const float* __restrict__ Jb,
                        float* __restrict__ Grel) {
  int bb = blockIdx.x * blockDim.x + threadIdx.x;
  if (bb >= BATCH) return;
  const int par[SMPL_NJ] = {0,0,0,0,1,2,3,4,5,6,7,8,9,9,9,12,13,14,16,17,18,19,20,21};
  float G[SMPL_NJ][12];
  for (int j = 0; j < SMPL_NJ; ++j) {
    const float* R  = Rb + ((long)bb * SMPL_NJ + j) * 9;
    const float* Jj = Jb + ((long)bb * SMPL_NJ + j) * 3;
    float t0, t1, t2;
    if (j == 0) { t0 = Jj[0]; t1 = Jj[1]; t2 = Jj[2]; }
    else {
      const float* Jp = Jb + ((long)bb * SMPL_NJ + par[j]) * 3;
      t0 = Jj[0]-Jp[0]; t1 = Jj[1]-Jp[1]; t2 = Jj[2]-Jp[2];
    }
    float A[12] = { R[0],R[1],R[2],t0, R[3],R[4],R[5],t1, R[6],R[7],R[8],t2 };
    if (j == 0) {
      for (int t = 0; t < 12; ++t) G[0][t] = A[t];
    } else {
      const float* P = G[par[j]];
      for (int r = 0; r < 3; ++r)
        for (int c = 0; c < 4; ++c)
          G[j][r*4+c] = P[r*4+0]*A[0*4+c] + P[r*4+1]*A[1*4+c] +
                        P[r*4+2]*A[2*4+c] + (c == 3 ? P[r*4+3] : 0.0f);
    }
  }
  for (int j = 0; j < SMPL_NJ; ++j) {
    const float* Jj = Jb + ((long)bb * SMPL_NJ + j) * 3;
    float* o = Grel + ((long)bb * SMPL_NJ + j) * 12;
    for (int r = 0; r < 3; ++r) {
      float gj = G[j][r*4+0]*Jj[0] + G[j][r*4+1]*Jj[1] + G[j][r*4+2]*Jj[2];
      o[r*4+0] = G[j][r*4+0]; o[r*4+1] = G[j][r*4+1];
      o[r*4+2] = G[j][r*4+2]; o[r*4+3] = G[j][r*4+3] - gj;
    }
  }
}

__global__ __launch_bounds__(256)
void k_skin(const float* __restrict__ posedirs, const float* __restrict__ wts,
            const float* __restrict__ vsh, const float* __restrict__ lrot,
            const float* __restrict__ Grel, const float* __restrict__ trans,
            float* __restrict__ verts) {
  __shared__ float sl[207];
  __shared__ float sg[288];
  int bb = blockIdx.y;
  for (int t = threadIdx.x; t < 207; t += 256) sl[t] = lrot[(long)bb * 207 + t];
  for (int t = threadIdx.x; t < 288; t += 256) sg[t] = Grel[(long)bb * 288 + t];
  __syncthreads();
  int v = blockIdx.x * 256 + threadIdx.x;
  if (v >= SMPL_V) return;
  float pv[4];
  for (int c = 0; c < 3; ++c) {
    float s = vsh[((long)bb * SMPL_V + v) * 3 + c];
    const float* pd = posedirs + ((long)v * 3 + c) * 207;
    for (int p = 0; p < 207; ++p) s += pd[p] * sl[p];
    pv[c] = s;
  }
  pv[3] = 1.0f;
  float T[12] = {0};
  for (int j = 0; j < SMPL_NJ; ++j) {
    float w = wts[(long)v * SMPL_NJ + j];
    const float* g = sg + j * 12;
    for (int t = 0; t < 12; ++t) T[t] += w * g[t];
  }
  float* o = verts + ((long)bb * SMPL_V + v) * 3;
  for (int c = 0; c < 3; ++c)
    o[c] = T[c*4+0]*pv[0] + T[c*4+1]*pv[1] + T[c*4+2]*pv[2] + T[c*4+3]
         + trans[bb * 3 + c];
}

__global__ void k_select(const int* __restrict__ genders,
                         const float* __restrict__ jf, const float* __restrict__ jm,
                         float* __restrict__ out) {
  int i = blockIdx.x * blockDim.x + threadIdx.x;
  if (i >= BATCH * 72) return;
  int bb = i / 72;
  out[i] = (genders[bb] == 0) ? jf[i] : jm[i];
}

// ---------------------------------------------------------------------------
// host orchestration
// ---------------------------------------------------------------------------
extern "C" void kernel_launch(void* const* d_in, const int* in_sizes, int n_in,
                              void* d_out, int out_size, void* d_ws, size_t ws_size,
                              hipStream_t stream) {
  (void)in_sizes; (void)n_in; (void)out_size; (void)ws_size;
  const int B = BATCH;

  // -------- input index map (depth-first dict-insertion-order flatten) ------
  int idx = 3;  // 0: x, 1: genders, 2: trans
  const int i_conv1 = idx++; const int i_bn1 = idx; idx += 4;
  struct Blk { int c1, b1, c2, b2, cd, bd; } blk[8];
  int bi = 0;
  for (int L = 0; L < 4; ++L)
    for (int j = 0; j < 2; ++j) {
      bool down = (L > 0 && j == 0);
      blk[bi].c1 = idx++; blk[bi].b1 = idx; idx += 4;
      blk[bi].c2 = idx++; blk[bi].b2 = idx; idx += 4;
      if (down) { blk[bi].cd = idx++; blk[bi].bd = idx; idx += 4; }
      else      { blk[bi].cd = -1;    blk[bi].bd = -1; }
      ++bi;
    }
  const int i_fcw = idx++, i_fcb = idx++;
  int i_smpl[2];
  i_smpl[0] = idx; idx += 5;
  i_smpl[1] = idx; idx += 5;

  // -------- conv sequence (network order): [w_idx, bn_idx, Ci, Co, k] ------
  int cW[20], cBN[20], cCi[20], cCo[20], cK[20]; int nconv = 0;
  auto addconv = [&](int w, int bn, int ci, int co, int k) {
    cW[nconv] = w; cBN[nconv] = bn; cCi[nconv] = ci; cCo[nconv] = co; cK[nconv] = k; ++nconv;
  };
  addconv(i_conv1, i_bn1, 1, 64, 7);
  {
    int chans[4] = {64, 128, 256, 512}; int inC = 64; int b2 = 0;
    for (int L = 0; L < 4; ++L)
      for (int j = 0; j < 2; ++j) {
        int outC = chans[L];
        addconv(blk[b2].c1, blk[b2].b1, inC, outC, 3);
        addconv(blk[b2].c2, blk[b2].b2, outC, outC, 3);
        if (blk[b2].cd >= 0) addconv(blk[b2].cd, blk[b2].bd, inC, outC, 1);
        inC = outC; ++b2;
      }
  }

  // -------- workspace carving ----------------------------------------------
  char* wsb = (char*)d_ws; size_t off = 0;
  auto alloc = [&](size_t bytes) -> void* {
    off = (off + 255) & ~(size_t)255;
    void* p = wsb + off; off += bytes; return p;
  };
  _Float16* x16 = (_Float16*)alloc((size_t)B * 224 * 224 * 2);
  _Float16* a1  = (_Float16*)alloc((size_t)B * 64 * 112 * 112 * 2);   // 102.8 MB
  size_t rbsz = (size_t)B * 64 * 56 * 56 * 2;                          // 25.69 MB
  // a1 is dead after maxpool -> reuse its region for 3 ping-pong buffers
  _Float16* rb0 = a1;
  _Float16* rb1 = (_Float16*)((char*)a1 + rbsz);
  _Float16* rb2 = (_Float16*)((char*)a1 + 2 * rbsz);
  _Float16* rb3 = (_Float16*)alloc(rbsz);                              // pool output

  _Float16* Wf16[20]; float* Sc[20]; float* Sh[20];
  for (int c = 0; c < nconv; ++c) {
    size_t nw = (size_t)cCo[c] * cCi[c] * cK[c] * cK[c];
    Wf16[c] = (_Float16*)alloc(nw * 2);
    Sc[c]   = (float*)alloc((size_t)cCo[c] * 4);
    Sh[c]   = (float*)alloc((size_t)cCo[c] * 4);
  }
  float* feat    = (float*)alloc((size_t)B * 512 * 4);
  float* feat82  = (float*)alloc((size_t)B * 82 * 4);
  float* Rbuf    = (float*)alloc((size_t)B * SMPL_NJ * 9 * 4);
  float* lrot    = (float*)alloc((size_t)B * 207 * 4);
  float* vsh     = (float*)alloc((size_t)B * SMPL_V * 3 * 4);
  float* verts   = (float*)alloc((size_t)B * SMPL_V * 3 * 4);
  float* Jbuf    = (float*)alloc((size_t)B * SMPL_NJ * 3 * 4);
  float* Grel    = (float*)alloc((size_t)B * SMPL_NJ * 12 * 4);
  float* jointsF = (float*)alloc((size_t)B * 72 * 4);
  float* jointsM = (float*)alloc((size_t)B * 72 * 4);

  // -------- prep: weights (pack for fast path), BN fold, input convert -----
  for (int c = 0; c < nconv; ++c) {
    int nw = cCo[c] * cCi[c] * cK[c] * cK[c];
    if (c == 0)   // conv1: generic kernel, flat OIHW f16
      k_f32_to_f16<<<(nw + 255) / 256, 256, 0, stream>>>(
          (const float*)d_in[cW[c]], Wf16[c], nw);
    else          // fast path: [tap][co][ci]
      k_pack_wgt<<<(nw + 255) / 256, 256, 0, stream>>>(
          (const float*)d_in[cW[c]], Wf16[c], cCo[c], cCi[c], cK[c], cK[c]);
    int b0 = cBN[c];
    k_bn_prep<<<(cCo[c] + 63) / 64, 64, 0, stream>>>(
        (const float*)d_in[b0], (const float*)d_in[b0 + 1],
        (const float*)d_in[b0 + 2], (const float*)d_in[b0 + 3],
        Sc[c], Sh[c], cCo[c]);
  }
  { int n = B * 224 * 224;   // NHWC with C=1 == NCHW
    k_f32_to_f16<<<(n + 255) / 256, 256, 0, stream>>>((const float*)d_in[0], x16, n); }

  // -------- backbone -------------------------------------------------------
  auto conv_fast = [&](const _Float16* in, int Ci, int Hin, int cidx,
                       const _Float16* res, _Float16* out, int stride, int pad,
                       int relu, int k) {
    int Co = cCo[cidx];
    int H = (Hin + 2 * pad - k) / stride + 1;
    int M = B * H * H;
    dim3 g(M / 32, Co / 64);
    k_conv_wmma_nhwc<<<g, 32, 0, stream>>>(in, Wf16[cidx], Sc[cidx], Sh[cidx], res, out,
                                           B, Ci, Hin, Hin, Co, H, H, k, k, stride, pad, relu);
  };

  // conv1 (7x7/2, Ci=1) via generic gather kernel -> NHWC 112x112x64
  {
    int H = 112, M = B * H * H, Co = 64;
    dim3 g((M + 15) / 16, (Co + 15) / 16);
    k_conv_wmma_gen<<<g, 32, 0, stream>>>(x16, Wf16[0], Sc[0], Sh[0], a1,
                                          B, 1, 224, 224, Co, H, H, 7, 7, 2, 3, 1);
  }
  { int n = B * 64 * 56 * 56;
    k_maxpool<<<(n + 255) / 256, 256, 0, stream>>>(a1, rb3, B, 64, 112, 112, 56, 56); }

  _Float16 *cur = rb3, *s0 = rb0, *s1 = rb1, *s2 = rb2;
  int curH = 56, curC = 64, cseq = 1;
  {
    int chans[4] = {64, 128, 256, 512};
    for (int L = 0; L < 4; ++L)
      for (int j = 0; j < 2; ++j) {
        int outC = chans[L];
        int stride = (L > 0 && j == 0) ? 2 : 1;
        bool down = (L > 0 && j == 0);
        int H2 = (curH + 2 - 3) / stride + 1;
        int c1 = cseq++, c2 = cseq++, cd = down ? cseq++ : -1;
        conv_fast(cur, curC, curH, c1, nullptr, s0, stride, 1, 1, 3);   // t1
        const _Float16* resp = cur;
        if (down) { conv_fast(cur, curC, curH, cd, nullptr, s1, stride, 0, 0, 1); resp = s1; }
        conv_fast(s0, outC, H2, c2, resp, s2, 1, 1, 1, 3);              // relu(bn+res)
        _Float16* t = cur; cur = s2; s2 = s1; s1 = s0; s0 = t;          // rotate
        curC = outC; curH = H2;
      }
  }

  { int n = B * 512;
    k_avgpool<<<(n + 255) / 256, 256, 0, stream>>>(cur, feat, B, 512, 49); }
  { int n = B * 82;
    k_fc<<<(n + 255) / 256, 256, 0, stream>>>(feat, (const float*)d_in[i_fcw],
                                              (const float*)d_in[i_fcb], feat82, B, 512, 82); }

  // -------- SMPL -----------------------------------------------------------
  { int n = B * SMPL_NJ;
    k_rodrigues<<<(n + 255) / 256, 256, 0, stream>>>(feat82, Rbuf, lrot); }

  const float* trans = (const float*)d_in[2];
  float* joints[2] = { jointsF, jointsM };
  for (int m = 0; m < 2; ++m) {
    int base = i_smpl[m];
    const float* shapedirs = (const float*)d_in[base + 0];
    const float* posedirs  = (const float*)d_in[base + 1];
    const float* v_templ   = (const float*)d_in[base + 2];
    const float* Jreg      = (const float*)d_in[base + 3];
    const float* weights   = (const float*)d_in[base + 4];

    { int n = B * SMPL_V;
      k_smpl_shape<<<(n + 255) / 256, 256, 0, stream>>>(shapedirs, v_templ, feat82, vsh); }
    k_jreg<<<B * SMPL_NJ, 256, 0, stream>>>(Jreg, vsh, Jbuf);
    k_chain<<<(B + 63) / 64, 64, 0, stream>>>(Rbuf, Jbuf, Grel);
    { dim3 g((SMPL_V + 255) / 256, B);
      k_skin<<<g, 256, 0, stream>>>(posedirs, weights, vsh, lrot, Grel, trans, verts); }
    k_jreg<<<B * SMPL_NJ, 256, 0, stream>>>(Jreg, verts, joints[m]);
  }

  { int n = B * 72;
    k_select<<<(n + 255) / 256, 256, 0, stream>>>((const int*)d_in[1],
                                                  jointsF, jointsM, (float*)d_out); }
}
